// MambaBlock_46926812676556
// MI455X (gfx1250) — matrix operand order repaired and checked by reading, not compiled, not run on hardware
//
#include <hip/hip_runtime.h>
#include <hip/hip_bf16.h>

// ---------------- problem constants ----------------
#define BB      4
#define LL      1024
#define DMODEL  512
#define DIN     1024
#define NST     16
#define RR      32
#define KC      4
#define MROWS   (BB * LL)          // 4096
#define DBLW    (RR + 2 * NST)     // 64

// ---------------- WMMA vector types ----------------
typedef __attribute__((ext_vector_type(16))) __bf16         v16bf;
typedef __attribute__((ext_vector_type(16))) unsigned short v16u;
typedef __attribute__((ext_vector_type(8)))  unsigned short v8u;
typedef __attribute__((ext_vector_type(8)))  float          v8f;

union BfVec {
    v16u u;
    v16bf h;
    struct { v8u lo, hi; } p;
};

// ---------------- helpers ----------------
__device__ __forceinline__ unsigned short f2bf(float f) {
    union { float f; unsigned int u; } v; v.f = f;
    unsigned int r = v.u + 0x7FFFu + ((v.u >> 16) & 1u);   // round-to-nearest-even
    return (unsigned short)(r >> 16);
}

__device__ __forceinline__ float silu_f(float x) {
    return x / (1.0f + __expf(-x));
}

__device__ __forceinline__ float softplus_f(float x) {
    return (x > 20.0f) ? x : __logf(1.0f + __expf(x));
}

// ---------------- f32 -> bf16 conversion (row-major copy) ----------------
__global__ void cvt_bf16_kernel(const float* __restrict__ src,
                                unsigned short* __restrict__ dst, int n) {
    int t = blockIdx.x * blockDim.x + threadIdx.x;
    if (t < n) dst[t] = f2bf(src[t]);
}

// ---------------- f32 -> bf16 conversion + transpose ----------------
// src[rows, cols] -> dst[cols, rows]   (weights: makes WMMA B-operand K-contiguous)
__global__ void cvt_bf16_t_kernel(const float* __restrict__ src,
                                  unsigned short* __restrict__ dst,
                                  int rows, int cols) {
    int t = blockIdx.x * blockDim.x + threadIdx.x;
    if (t >= rows * cols) return;
    const int r = t / cols;
    const int c = t - r * cols;
    dst[(size_t)c * rows + r] = f2bf(src[t]);
}

// ---------------- bf16 WMMA GEMM, B pre-transposed ----------------
// C[M,N] = A[M,K] * Bt[N,K]^T   (A row-major M*K, Bt row-major N*K, C f32 row-major)
// Each wave computes a 16x64 tile (4 accumulators, A fragment reused 4x).
// Per k-step: A = 2x b128 loads, each B tile = 2x b128 loads.
// epilogue: 0 = none, 1 = +bias[col], 2 = softplus(x + bias[col])
__global__ void gemm_bf16_wmma(const unsigned short* __restrict__ A,
                               const unsigned short* __restrict__ Bt,
                               float* __restrict__ C,
                               int M, int N, int K,
                               int epilogue, const float* __restrict__ bias) {
    const int gtid  = blockIdx.x * blockDim.x + threadIdx.x;
    const int wave  = gtid >> 5;
    const int lane  = threadIdx.x & 31;
    const int ntile = N >> 6;                 // 64-wide column tiles
    const int tm    = wave / ntile;
    const int tn    = wave - tm * ntile;
    if (tm >= (M >> 4)) return;

    const int row0 = tm << 4;
    const int col0 = tn << 6;

    // A operand: lane&15 = M row; lanes 0-15 K base 0, lanes 16-31 K base 16
    //   per-lane K = {kb..kb+15}  -> two contiguous 8xbf16 chunks
    const int kbaseA = (lane < 16) ? 0 : 16;
    const unsigned short* Arow = A + (size_t)(row0 + (lane & 15)) * K;

    // B operand: lane&15 = N col; lanes 0-15 K base 0, lanes 16-31 K base 16
    //   per-lane K = {kb..kb+15}  -> one contiguous 16xbf16 run in Bt
    const int kbaseB = (lane < 16) ? 0 : 16;
    const unsigned short* Brow[4];
#pragma unroll
    for (int t = 0; t < 4; ++t)
        Brow[t] = Bt + (size_t)(col0 + 16 * t + (lane & 15)) * K;

    v8f acc0 = {}, acc1 = {}, acc2 = {}, acc3 = {};

    for (int k0 = 0; k0 < K; k0 += 32) {
        BfVec a;
        a.p.lo = *(const v8u*)(Arow + k0 + kbaseA);        // K = kb .. kb+7
        a.p.hi = *(const v8u*)(Arow + k0 + kbaseA + 8);    // K = kb+8 .. kb+15

        BfVec b0, b1, b2, b3;
        b0.p.lo = *(const v8u*)(Brow[0] + k0 + kbaseB);
        b0.p.hi = *(const v8u*)(Brow[0] + k0 + kbaseB + 8);
        b1.p.lo = *(const v8u*)(Brow[1] + k0 + kbaseB);
        b1.p.hi = *(const v8u*)(Brow[1] + k0 + kbaseB + 8);
        b2.p.lo = *(const v8u*)(Brow[2] + k0 + kbaseB);
        b2.p.hi = *(const v8u*)(Brow[2] + k0 + kbaseB + 8);
        b3.p.lo = *(const v8u*)(Brow[3] + k0 + kbaseB);
        b3.p.hi = *(const v8u*)(Brow[3] + k0 + kbaseB + 8);

        acc0 = __builtin_amdgcn_wmma_f32_16x16x32_bf16(false, a.h, false, b0.h, (short)0, acc0, false, false);
        acc1 = __builtin_amdgcn_wmma_f32_16x16x32_bf16(false, a.h, false, b1.h, (short)0, acc1, false, false);
        acc2 = __builtin_amdgcn_wmma_f32_16x16x32_bf16(false, a.h, false, b2.h, (short)0, acc2, false, false);
        acc3 = __builtin_amdgcn_wmma_f32_16x16x32_bf16(false, a.h, false, b3.h, (short)0, acc3, false, false);
    }

    // D layout: VGPR r -> rows (r | r+8), lanes 0-15 / 16-31, col = lane&15
    const int crow0 = row0 + ((lane < 16) ? 0 : 8);
    v8f accs[4] = {acc0, acc1, acc2, acc3};
#pragma unroll
    for (int t = 0; t < 4; ++t) {
        const int ccol = col0 + 16 * t + (lane & 15);
        const float bia = (epilogue != 0) ? bias[ccol] : 0.0f;
#pragma unroll
        for (int r = 0; r < 8; ++r) {
            float v = accs[t][r] + bia;
            if (epilogue == 2) v = softplus_f(v);
            C[(size_t)(crow0 + r) * N + ccol] = v;
        }
    }
}

// ---------------- depthwise causal conv (K=4) + SiLU ----------------
// xres: [MROWS, 2*DIN]; xs occupies cols [0,DIN)
__global__ void conv_silu_kernel(const float* __restrict__ xres,
                                 const float* __restrict__ conv_w,
                                 const float* __restrict__ conv_b,
                                 float* __restrict__ xs_act,
                                 unsigned short* __restrict__ xs_act_bf) {
    int t = blockIdx.x * blockDim.x + threadIdx.x;
    if (t >= MROWS * DIN) return;
    const int d  = t & (DIN - 1);
    const int l  = (t >> 10) & (LL - 1);
    const int b  = t >> 20;

    float acc = conv_b[d];
#pragma unroll
    for (int k = 0; k < KC; ++k) {
        const int ls = l + k - (KC - 1);
        if (ls >= 0)
            acc += conv_w[d * KC + k] * xres[(size_t)(b * LL + ls) * (2 * DIN) + d];
    }
    const float s = silu_f(acc);
    xs_act[t]    = s;
    xs_act_bf[t] = f2bf(s);
}

// ---------------- extract delta-projection input (first R cols of x_dbl) ----
__global__ void extract_dtin_kernel(const float* __restrict__ x_dbl,
                                    unsigned short* __restrict__ dt_in) {
    int t = blockIdx.x * blockDim.x + threadIdx.x;
    if (t >= MROWS * RR) return;
    const int row = t / RR;
    const int c   = t - row * RR;
    dt_in[t] = f2bf(x_dbl[(size_t)row * DBLW + c]);
}

// ---------------- selective scan: one thread per (b,d) ----------------
__global__ void scan_kernel(const float* __restrict__ xs_act,
                            const float* __restrict__ delta,
                            const float* __restrict__ x_dbl,
                            const float* __restrict__ A_log,
                            const float* __restrict__ Dvec,
                            float* __restrict__ y) {
    int t = blockIdx.x * blockDim.x + threadIdx.x;
    if (t >= BB * DIN) return;
    const int b = t / DIN;
    const int d = t - b * DIN;

    float Arow[NST];
#pragma unroll
    for (int n = 0; n < NST; ++n) Arow[n] = -__expf(A_log[d * NST + n]);
    const float Dd = Dvec[d];

    float s[NST];
#pragma unroll
    for (int n = 0; n < NST; ++n) s[n] = 0.0f;

    for (int l = 0; l < LL; ++l) {
        const size_t idx = (size_t)(b * LL + l);
        const float u  = xs_act[idx * DIN + d];
        const float dt = delta [idx * DIN + d];
        const float du = dt * u;
        const float* bc = x_dbl + idx * DBLW;
        float acc = 0.0f;
#pragma unroll
        for (int n = 0; n < NST; ++n) {
            const float a = (l == 0) ? 1.0f : __expf(dt * Arow[n]);   // dA_shift semantics
            s[n] = a * s[n] + du * bc[RR + n];
            acc += s[n] * bc[RR + NST + n];
        }
        y[idx * DIN + d] = acc + u * Dd;
    }
}

// ---------------- gate: y * silu(res) -> bf16 ----------------
__global__ void gate_kernel(const float* __restrict__ y,
                            const float* __restrict__ xres,
                            unsigned short* __restrict__ yg_bf) {
    int t = blockIdx.x * blockDim.x + threadIdx.x;
    if (t >= MROWS * DIN) return;
    const int row = t >> 10;
    const int d   = t & (DIN - 1);
    const float res = xres[(size_t)row * (2 * DIN) + DIN + d];
    yg_bf[t] = f2bf(y[t] * silu_f(res));
}

// ---------------- launch ----------------
extern "C" void kernel_launch(void* const* d_in, const int* in_sizes, int n_in,
                              void* d_out, int out_size, void* d_ws, size_t ws_size,
                              hipStream_t stream) {
    const float* x      = (const float*)d_in[0];
    const float* W_in   = (const float*)d_in[1];
    const float* conv_w = (const float*)d_in[2];
    const float* conv_b = (const float*)d_in[3];
    const float* W_x    = (const float*)d_in[4];
    const float* W_dt   = (const float*)d_in[5];
    const float* b_dt   = (const float*)d_in[6];
    const float* A_log  = (const float*)d_in[7];
    const float* Dvec   = (const float*)d_in[8];
    const float* W_out  = (const float*)d_in[9];
    const float* b_out  = (const float*)d_in[10];
    float* out = (float*)d_out;

    // ---- workspace carve-up (all offsets 256B aligned) ----
    char* ws = (char*)d_ws;
    size_t off = 0;
    auto alloc = [&](size_t bytes) { char* p = ws + off; off += (bytes + 255) & ~(size_t)255; return p; };

    unsigned short* xbf      = (unsigned short*)alloc((size_t)MROWS * DMODEL * 2);
    unsigned short* winbf_t  = (unsigned short*)alloc((size_t)(2 * DIN) * DMODEL * 2); // [2048,512]
    unsigned short* wxbf_t   = (unsigned short*)alloc((size_t)DBLW * DIN * 2);         // [64,1024]
    unsigned short* wdtbf_t  = (unsigned short*)alloc((size_t)DIN * RR * 2);           // [1024,32]
    unsigned short* woutbf_t = (unsigned short*)alloc((size_t)DMODEL * DIN * 2);       // [512,1024]
    float*          xres     = (float*)alloc((size_t)MROWS * 2 * DIN * 4);
    float*          xs_act   = (float*)alloc((size_t)MROWS * DIN * 4);
    unsigned short* xsact_bf = (unsigned short*)alloc((size_t)MROWS * DIN * 2);
    float*          x_dbl    = (float*)alloc((size_t)MROWS * DBLW * 4);
    unsigned short* dtin_bf  = (unsigned short*)alloc((size_t)MROWS * RR * 2);
    float*          delta    = (float*)alloc((size_t)MROWS * DIN * 4);
    float*          yscan    = (float*)alloc((size_t)MROWS * DIN * 4);
    unsigned short* yg_bf    = (unsigned short*)alloc((size_t)MROWS * DIN * 2);
    (void)ws_size; (void)n_in; (void)in_sizes; (void)out_size;

    const int TB = 256;

    // 1) bf16 copies: activations row-major, weights transposed (K-contiguous per output col)
    cvt_bf16_kernel<<<(MROWS * DMODEL + TB - 1) / TB, TB, 0, stream>>>(x, xbf, MROWS * DMODEL);
    cvt_bf16_t_kernel<<<(DMODEL * 2 * DIN + TB - 1) / TB, TB, 0, stream>>>(W_in,  winbf_t,  DMODEL, 2 * DIN);
    cvt_bf16_t_kernel<<<(DIN * DBLW     + TB - 1) / TB, TB, 0, stream>>>(W_x,   wxbf_t,   DIN, DBLW);
    cvt_bf16_t_kernel<<<(RR * DIN       + TB - 1) / TB, TB, 0, stream>>>(W_dt,  wdtbf_t,  RR, DIN);
    cvt_bf16_t_kernel<<<(DIN * DMODEL   + TB - 1) / TB, TB, 0, stream>>>(W_out, woutbf_t, DIN, DMODEL);

    auto gemm = [&](const unsigned short* A, const unsigned short* Bt, float* C,
                    int M, int N, int K, int ep, const float* bias) {
        const int waves   = (M >> 4) * (N >> 6);   // 16x64 tile per wave
        const int threads = waves * 32;
        gemm_bf16_wmma<<<(threads + 127) / 128, 128, 0, stream>>>(A, Bt, C, M, N, K, ep, bias);
    };

    // 2) x_and_res = x @ W_in          (4096x512)(512x2048)
    gemm(xbf, winbf_t, xres, MROWS, 2 * DIN, DMODEL, 0, nullptr);

    // 3) depthwise causal conv + SiLU
    conv_silu_kernel<<<(MROWS * DIN + TB - 1) / TB, TB, 0, stream>>>(
        xres, conv_w, conv_b, xs_act, xsact_bf);

    // 4) x_dbl = xs @ W_x              (4096x1024)(1024x64)
    gemm(xsact_bf, wxbf_t, x_dbl, MROWS, DBLW, DIN, 0, nullptr);

    // 5) dt_in = bf16(x_dbl[:, :R])
    extract_dtin_kernel<<<(MROWS * RR + TB - 1) / TB, TB, 0, stream>>>(x_dbl, dtin_bf);

    // 6) delta = softplus(dt_in @ W_dt + b_dt)   (4096x32)(32x1024)
    gemm(dtin_bf, wdtbf_t, delta, MROWS, DIN, RR, 2, b_dt);

    // 7) selective scan (fp32 state in registers, one thread per (b,d))
    scan_kernel<<<(BB * DIN + 127) / 128, 128, 0, stream>>>(
        xs_act, delta, x_dbl, A_log, Dvec, yscan);

    // 8) gate with silu(res) -> bf16
    gate_kernel<<<(MROWS * DIN + TB - 1) / TB, TB, 0, stream>>>(yscan, xres, yg_bf);

    // 9) out = y_g @ W_out + b_out     (4096x1024)(1024x512)
    gemm(yg_bf, woutbf_t, out, MROWS, DMODEL, DIN, 1, b_out);
}
